// StandardHeteroGNN_1099511628112
// MI455X (gfx1250) — compile-verified, matrix operand order checked by reading, not run on hardware
//
#include <hip/hip_runtime.h>

// ---------------------------------------------------------------------------
// CDNA5 (gfx1250) HeteroGNN: fused SAGE scatter + WMMA GEMM + LN/ReLU + MLP
// - Weights pre-converted to f16 once per layer (self weights pre-summed).
// - Activations/means converted to f16 cooperatively into LDS once per block
//   (scale folded in), then all 8 waves build WMMA fragments from LDS.
// ---------------------------------------------------------------------------

typedef __attribute__((ext_vector_type(16))) _Float16 v16h;
typedef __attribute__((ext_vector_type(8)))  _Float16 v8h;
typedef __attribute__((ext_vector_type(8)))  float    v8f;

#define HDIM 128

// ---------------------------------------------------------------------------
// Fragment loader for v_wmma_f32_16x16x32_f16 (wave32).
// 16-bit A-matrix 16x32 layout: lane L holds row M = L&15; with hi = L>>4,
//   halves e=0..7  -> K = hi*8 + e
//   halves e=8..15 -> K = 16 + hi*8 + (e-8)
// B (32x16) mirrors this with N in place of M: lane holds W[n, k] with the
// identical K mapping, computing out = A @ W^T.
// Works for both global (weights) and LDS (staged activations) pointers.
// ---------------------------------------------------------------------------
__device__ __forceinline__ v16h frag_ld_f16(const _Float16* p, int hi) {
  const v8h lo = *(const v8h*)(p + hi * 8);
  const v8h hh = *(const v8h*)(p + 16 + hi * 8);
  return __builtin_shufflevector(lo, hh, 0, 1, 2, 3, 4, 5, 6, 7,
                                 8, 9, 10, 11, 12, 13, 14, 15);
}

// ---------------------------------------------------------------------------
// Weight prep: out = f16(a [+ b]); also folds Wr_rev + Wr_fol.
// ---------------------------------------------------------------------------
__global__ __launch_bounds__(256)
void wconv_kernel(const float* __restrict__ a, const float* __restrict__ b,
                  _Float16* __restrict__ out, int n) {
  const int i = blockIdx.x * blockDim.x + threadIdx.x;
  if (i < n) out[i] = (_Float16)(b ? (a[i] + b[i]) : a[i]);
}

// ---------------------------------------------------------------------------
// Edge scatter: one wave per edge. Lane-parallel float4 gather of the source
// feature row, 4 hardware global f32 atomic adds per lane into agg[dst],
// lane0 bumps the count. Atomics resolve in L2 (192MB holds the 51MB tables).
// ---------------------------------------------------------------------------
__global__ __launch_bounds__(256)
void scatter_kernel(const float* __restrict__ xsrc,
                    const int* __restrict__ src, const int* __restrict__ dst,
                    float* __restrict__ agg, float* __restrict__ cnt, int nedges) {
  const int e = blockIdx.x * (blockDim.x >> 5) + (threadIdx.x >> 5);
  const int lane = threadIdx.x & 31;
  if (e >= nedges) return;
  const int s = src[e];
  const int d = dst[e];
  const float4 v = *(const float4*)(xsrc + (size_t)s * HDIM + lane * 4);
  float* ap = agg + (size_t)d * HDIM + lane * 4;
  unsafeAtomicAdd(ap + 0, v.x);
  unsafeAtomicAdd(ap + 1, v.y);
  unsafeAtomicAdd(ap + 2, v.z);
  unsafeAtomicAdd(ap + 3, v.w);
  if (lane == 0) unsafeAtomicAdd(cnt + d, 1.0f);
}

// ---------------------------------------------------------------------------
// Fused SAGE linear + bias + LayerNorm + ReLU.
//   out = relu(LN( mean_A @ WlA^T + [mean_B @ WlB^T] + x @ WrS^T + bias ))
// Block = 256 threads = 8 waves, owns a 16-row x 128-col output tile.
// Phase 1: cooperatively stage the (2 or 3) A-source 16x128 tiles into LDS
//          as f16 with the mean scale folded in (one global read each).
// Phase 2: wave w computes column tile [16w,16w+16) with 8-12 WMMAs; A-frags
//          come from LDS (ds_load_b128 pairs), B-frags from f16 weights.
// Phase 3: f32 results (+bias) parked in a 16x128 LDS tile; cross-wave
//          LayerNorm via wave shuffles; ReLU; streamed out.
// ---------------------------------------------------------------------------
__global__ __launch_bounds__(256)
void sage_gemm_ln_kernel(const float* __restrict__ xdst,
                         const float* __restrict__ aggA, const float* __restrict__ cntA,
                         const float* __restrict__ aggB, const float* __restrict__ cntB,
                         const _Float16* __restrict__ wlA16,
                         const _Float16* __restrict__ wlB16,   // null => single-source
                         const _Float16* __restrict__ wrS16,   // pre-summed self weights
                         const float* __restrict__ blA, const float* __restrict__ blB,
                         const float* __restrict__ ln_g, const float* __restrict__ ln_b,
                         float* __restrict__ out, int nrows) {
  __shared__ _Float16 aT[3 * 16 * HDIM];   // staged f16 A-sources (12KB)
  __shared__ float    tile[16 * HDIM];     // f32 results for LN (8KB)
  __shared__ float    sInv[2 * 16];        // per-row mean reciprocals

  const int tid  = threadIdx.x;
  const int wave = tid >> 5;               // 0..7 -> column tile
  const int lane = tid & 31;
  const int m16  = lane & 15;
  const int hi   = lane >> 4;
  const int nsrc = wlB16 ? 3 : 2;

  // Per-row 1/max(cnt,1) once per block.
  if (tid < 16) {
    int rg = blockIdx.x * 16 + tid; if (rg >= nrows) rg = nrows - 1;
    sInv[tid] = 1.0f / fmaxf(cntA[rg], 1.0f);
  } else if (tid < 32 && aggB) {
    const int r = tid - 16;
    int rg = blockIdx.x * 16 + r; if (rg >= nrows) rg = nrows - 1;
    sInv[16 + r] = 1.0f / fmaxf(cntB[rg], 1.0f);
  }
  __syncthreads();

  // Phase 1: stage A-sources. Each iteration handles one full source (256
  // groups of 8 floats = 16 rows x 128 cols); s is uniform per iteration.
  for (int s = 0; s < nsrc; ++s) {
    const int r = tid >> 4;                // row 0..15
    const int c = (tid & 15) * 8;          // col group
    int rg = blockIdx.x * 16 + r; if (rg >= nrows) rg = nrows - 1;
    const float* src;
    float scale;
    if (s == nsrc - 1)      { src = xdst; scale = 1.0f; }
    else if (s == 0)        { src = aggA; scale = sInv[r]; }
    else                    { src = aggB; scale = sInv[16 + r]; }
    const float4 v0 = *(const float4*)(src + (size_t)rg * HDIM + c);
    const float4 v1 = *(const float4*)(src + (size_t)rg * HDIM + c + 4);
    v8h h;
    h[0] = (_Float16)(v0.x * scale); h[1] = (_Float16)(v0.y * scale);
    h[2] = (_Float16)(v0.z * scale); h[3] = (_Float16)(v0.w * scale);
    h[4] = (_Float16)(v1.x * scale); h[5] = (_Float16)(v1.y * scale);
    h[6] = (_Float16)(v1.z * scale); h[7] = (_Float16)(v1.w * scale);
    *(v8h*)&aT[s * (16 * HDIM) + r * HDIM + c] = h;
  }
  __syncthreads();

  // Phase 2: WMMA. Each wave reads shared A-frags from LDS, own B-frags.
  const int n = wave * 16 + m16;           // output feature column
  const _Float16* tA = &aT[0 * (16 * HDIM) + m16 * HDIM];
  const _Float16* tB = &aT[1 * (16 * HDIM) + m16 * HDIM];          // if nsrc==3
  const _Float16* tX = &aT[(nsrc - 1) * (16 * HDIM) + m16 * HDIM];
  const _Float16* wlA = wlA16 + (size_t)n * HDIM;
  const _Float16* wlB = wlB16 ? wlB16 + (size_t)n * HDIM : nullptr;
  const _Float16* wrS = wrS16 + (size_t)n * HDIM;

  v8f acc = {};
#pragma unroll
  for (int ks = 0; ks < 4; ++ks) {
    const int koff = ks * 32;
    v16h aA = frag_ld_f16(tA + koff, hi);
    v16h bA = frag_ld_f16(wlA + koff, hi);
    acc = __builtin_amdgcn_wmma_f32_16x16x32_f16(false, aA, false, bA,
                                                 (short)0, acc, false, false);
    if (wlB) {
      v16h aB = frag_ld_f16(tB + koff, hi);
      v16h bB = frag_ld_f16(wlB + koff, hi);
      acc = __builtin_amdgcn_wmma_f32_16x16x32_f16(false, aB, false, bB,
                                                   (short)0, acc, false, false);
    }
    v16h aS = frag_ld_f16(tX + koff, hi);
    v16h bS = frag_ld_f16(wrS + koff, hi);
    acc = __builtin_amdgcn_wmma_f32_16x16x32_f16(false, aS, false, bS,
                                                 (short)0, acc, false, false);
  }

  const float bias = blA[n] + (blB ? blB[n] : 0.0f);

  // Phase 3: D layout -> LDS. Lane covers rows m = r + 8*hi (r=0..7), col n.
#pragma unroll
  for (int r = 0; r < 8; ++r)
    tile[(r + 8 * hi) * HDIM + n] = acc[r] + bias;

  __syncthreads();

  // LayerNorm + ReLU: wave w owns rows 2w and 2w+1; 32 lanes x float4 = 128.
#pragma unroll
  for (int rr = 2 * wave; rr < 2 * wave + 2; ++rr) {
    const float4 v = *(const float4*)&tile[rr * HDIM + lane * 4];
    float s  = v.x + v.y + v.z + v.w;
    float sq = v.x * v.x + v.y * v.y + v.z * v.z + v.w * v.w;
#pragma unroll
    for (int m = 16; m >= 1; m >>= 1) {
      s  += __shfl_xor(s,  m, 32);
      sq += __shfl_xor(sq, m, 32);
    }
    const float mu   = s * (1.0f / HDIM);
    const float var  = sq * (1.0f / HDIM) - mu * mu;
    const float rstd = rsqrtf(var + 1e-5f);
    const int col = lane * 4;
    const float4 g = *(const float4*)&ln_g[col];
    const float4 b = *(const float4*)&ln_b[col];
    float4 o;
    o.x = fmaxf((v.x - mu) * rstd * g.x + b.x, 0.0f);
    o.y = fmaxf((v.y - mu) * rstd * g.y + b.y, 0.0f);
    o.z = fmaxf((v.z - mu) * rstd * g.z + b.z, 0.0f);
    o.w = fmaxf((v.w - mu) * rstd * g.w + b.w, 0.0f);
    const int grow = blockIdx.x * 16 + rr;
    if (grow < nrows)
      *(float4*)&out[(size_t)grow * HDIM + col] = o;
  }
}

// ---------------------------------------------------------------------------
// Final MLP head: h = relu(x @ W1^T + b1); out = h @ W2^T + b2.
// One 64-thread block per target; thread j owns hidden unit j.
// ---------------------------------------------------------------------------
__global__ __launch_bounds__(64)
void mlp_kernel(const float* __restrict__ xu, const int* __restrict__ tgt,
                const float* __restrict__ W1, const float* __restrict__ b1,
                const float* __restrict__ W2, const float* __restrict__ b2,
                float* __restrict__ out, int ntgt) {
  __shared__ float red[64];
  const int t = blockIdx.x;
  if (t >= ntgt) return;
  const int j = threadIdx.x;
  const float* x = xu + (size_t)tgt[t] * HDIM;
  const float* w = W1 + (size_t)j * HDIM;
  float acc = b1[j];
#pragma unroll
  for (int k = 0; k < HDIM; k += 4) {
    const float4 xv = *(const float4*)(x + k);
    const float4 wv = *(const float4*)(w + k);
    acc = fmaf(xv.x, wv.x, acc);
    acc = fmaf(xv.y, wv.y, acc);
    acc = fmaf(xv.z, wv.z, acc);
    acc = fmaf(xv.w, wv.w, acc);
  }
  red[j] = fmaxf(acc, 0.0f) * W2[j];
  __syncthreads();
#pragma unroll
  for (int sft = 32; sft >= 1; sft >>= 1) {
    if (j < sft) red[j] += red[j + sft];
    __syncthreads();
  }
  if (j == 0) out[t] = red[0] + b2[0];
}

// ---------------------------------------------------------------------------
// Host orchestration
// ---------------------------------------------------------------------------
extern "C" void kernel_launch(void* const* d_in, const int* in_sizes, int n_in,
                              void* d_out, int out_size, void* d_ws, size_t ws_size,
                              hipStream_t stream) {
  const float* emb_user = (const float*)d_in[0];
  const float* emb_item = (const float*)d_in[1];
  const float* Wl   = (const float*)d_in[2];   // (2,3,128,128)
  const float* bl   = (const float*)d_in[3];   // (2,3,128)
  const float* Wr   = (const float*)d_in[4];   // (2,3,128,128)
  const float* ln_g = (const float*)d_in[5];   // (2,2,128)
  const float* ln_b = (const float*)d_in[6];   // (2,2,128)
  const float* W1   = (const float*)d_in[7];
  const float* b1   = (const float*)d_in[8];
  const float* W2   = (const float*)d_in[9];
  const float* b2   = (const float*)d_in[10];
  const int* src_buys = (const int*)d_in[11];
  const int* dst_buys = (const int*)d_in[12];
  const int* src_rev  = (const int*)d_in[13];
  const int* dst_rev  = (const int*)d_in[14];
  const int* src_fol  = (const int*)d_in[15];
  const int* dst_fol  = (const int*)d_in[16];
  const int* target_ids = (const int*)d_in[17];

  const int N_USER = in_sizes[0] / HDIM;
  const int N_ITEM = in_sizes[1] / HDIM;
  const int E      = in_sizes[11];
  const int N_TGT  = in_sizes[17];

  // Workspace carve-up (256B aligned)
  char* ws = (char*)d_ws;
  size_t off = 0;
  auto carve = [&](size_t bytes) -> char* {
    char* p = ws + off;
    off += (bytes + 255) & ~(size_t)255;
    return p;
  };
  const size_t ubytes = (size_t)N_USER * HDIM * sizeof(float);
  const size_t ibytes = (size_t)N_ITEM * HDIM * sizeof(float);
  const size_t WSZ    = (size_t)HDIM * HDIM;      // elements per weight matrix
  float* xu0      = (float*)carve(ubytes);
  float* xu1      = (float*)carve(ubytes);
  float* xi0      = (float*)carve(ibytes);
  float* xi1      = (float*)carve(ibytes);
  float* agg_rev  = (float*)carve(ubytes);
  float* agg_fol  = (float*)carve(ubytes);
  float* agg_buys = (float*)carve(ibytes);
  float* cnt_rev  = (float*)carve((size_t)N_USER * sizeof(float));
  float* cnt_fol  = (float*)carve((size_t)N_USER * sizeof(float));
  float* cnt_buys = (float*)carve((size_t)N_ITEM * sizeof(float));
  _Float16* wl0h  = (_Float16*)carve(WSZ * sizeof(_Float16));  // item lin_l
  _Float16* wr0h  = (_Float16*)carve(WSZ * sizeof(_Float16));  // item lin_r
  _Float16* wl1h  = (_Float16*)carve(WSZ * sizeof(_Float16));  // user rev lin_l
  _Float16* wl2h  = (_Float16*)carve(WSZ * sizeof(_Float16));  // user fol lin_l
  _Float16* wrSh  = (_Float16*)carve(WSZ * sizeof(_Float16));  // user lin_r sum

  const dim3 blk256(256), blk64(64);
  const int scatter_grid = (E + 7) / 8;                 // 8 edges (waves) / block
  const int user_grid = (N_USER + 15) / 16;
  const int item_grid = (N_ITEM + 15) / 16;
  const int wconv_grid = (int)(WSZ / 256);

  const float* xu_cur = emb_user;
  const float* xi_cur = emb_item;
  float* xu_bufs[2] = {xu0, xu1};
  float* xi_bufs[2] = {xi0, xi1};

  for (int l = 0; l < 2; ++l) {
    hipMemsetAsync(agg_rev,  0, ubytes, stream);
    hipMemsetAsync(agg_fol,  0, ubytes, stream);
    hipMemsetAsync(agg_buys, 0, ibytes, stream);
    hipMemsetAsync(cnt_rev,  0, (size_t)N_USER * sizeof(float), stream);
    hipMemsetAsync(cnt_fol,  0, (size_t)N_USER * sizeof(float), stream);
    hipMemsetAsync(cnt_buys, 0, (size_t)N_ITEM * sizeof(float), stream);

    scatter_kernel<<<scatter_grid, blk256, 0, stream>>>(
        xu_cur, src_buys, dst_buys, agg_buys, cnt_buys, E);
    scatter_kernel<<<scatter_grid, blk256, 0, stream>>>(
        xi_cur, src_rev, dst_rev, agg_rev, cnt_rev, E);
    scatter_kernel<<<scatter_grid, blk256, 0, stream>>>(
        xu_cur, src_fol, dst_fol, agg_fol, cnt_fol, E);

    const float* Wl0 = Wl + ((size_t)l * 3 + 0) * WSZ;  // buys
    const float* Wl1 = Wl + ((size_t)l * 3 + 1) * WSZ;  // rev_buys
    const float* Wl2 = Wl + ((size_t)l * 3 + 2) * WSZ;  // follows
    const float* Wr0 = Wr + ((size_t)l * 3 + 0) * WSZ;
    const float* Wr1 = Wr + ((size_t)l * 3 + 1) * WSZ;
    const float* Wr2 = Wr + ((size_t)l * 3 + 2) * WSZ;
    const float* bl0 = bl + ((size_t)l * 3 + 0) * HDIM;
    const float* bl1 = bl + ((size_t)l * 3 + 1) * HDIM;
    const float* bl2 = bl + ((size_t)l * 3 + 2) * HDIM;
    const float* gU = ln_g + ((size_t)l * 2 + 0) * HDIM;
    const float* bU = ln_b + ((size_t)l * 2 + 0) * HDIM;
    const float* gI = ln_g + ((size_t)l * 2 + 1) * HDIM;
    const float* bI = ln_b + ((size_t)l * 2 + 1) * HDIM;

    // Per-layer f16 weight prep (and fold Wr1+Wr2 for the user path).
    wconv_kernel<<<wconv_grid, blk256, 0, stream>>>(Wl0, nullptr, wl0h, (int)WSZ);
    wconv_kernel<<<wconv_grid, blk256, 0, stream>>>(Wr0, nullptr, wr0h, (int)WSZ);
    wconv_kernel<<<wconv_grid, blk256, 0, stream>>>(Wl1, nullptr, wl1h, (int)WSZ);
    wconv_kernel<<<wconv_grid, blk256, 0, stream>>>(Wl2, nullptr, wl2h, (int)WSZ);
    wconv_kernel<<<wconv_grid, blk256, 0, stream>>>(Wr1, Wr2,     wrSh, (int)WSZ);

    // item: buys conv only (self weights = Wr0 alone)
    sage_gemm_ln_kernel<<<item_grid, blk256, 0, stream>>>(
        xi_cur, agg_buys, cnt_buys, nullptr, nullptr,
        wl0h, nullptr, wr0h, bl0, nullptr,
        gI, bI, xi_bufs[l], N_ITEM);

    // user: rev_buys + follows convs summed
    sage_gemm_ln_kernel<<<user_grid, blk256, 0, stream>>>(
        xu_cur, agg_rev, cnt_rev, agg_fol, cnt_fol,
        wl1h, wl2h, wrSh, bl1, bl2,
        gU, bU, xu_bufs[l], N_USER);

    xu_cur = xu_bufs[l];
    xi_cur = xi_bufs[l];
  }

  mlp_kernel<<<N_TGT, blk64, 0, stream>>>(
      xu_cur, target_ids, W1, b1, W2, b2, (float*)d_out, N_TGT);
}
